// LoraLayer_62380105007111
// MI455X (gfx1250) — matrix operand order repaired and checked by reading, not compiled, not run on hardware
//
#include <hip/hip_runtime.h>

#define FEAT_IN  4096
#define FEAT_OUT 4096
#define M_TOTAL  8192
#define BM 128
#define BN 128
#define BK 32
#define NKSTEPS (FEAT_IN / BK)
#define XPITCH 40   // padded LDS row pitch in ushorts (80B, 16B-aligned, conflict-free)

typedef __attribute__((ext_vector_type(16))) __bf16        v16bf;
typedef __attribute__((ext_vector_type(8)))  float         v8f;
typedef __attribute__((ext_vector_type(4)))  unsigned int  v4u;
typedef __attribute__((ext_vector_type(8)))  int           v8i;
typedef __attribute__((ext_vector_type(4)))  int           v4i;

union BFrag { v16bf v; uint4 q[2]; };

__device__ __forceinline__ unsigned short f2bf(float f) {
    unsigned u = __float_as_uint(f);
    unsigned r = u + 0x7FFFu + ((u >> 16) & 1u);   // round-to-nearest-even
    return (unsigned short)(r >> 16);
}
__device__ __forceinline__ float bf2f(unsigned short h) {
    return __uint_as_float(((unsigned)h) << 16);
}

// ---------------------------------------------------------------------------
// Shared wave-level compute: 4(M) x 2(N) tiles of 16x16, 3-product split-bf16.
// ---------------------------------------------------------------------------
__device__ __forceinline__ void wave_mma_step(const unsigned short* xhi, const unsigned short* xlo,
                                              const unsigned short* whi, const unsigned short* wlo,
                                              int mOff, int nOff, int lrow, int halfsel,
                                              v8f (&acc)[4][2]) {
    BFrag bh[2], bl[2];
#pragma unroll
    for (int ns = 0; ns < 2; ++ns) {
        const int nrow = nOff + ns * 16 + lrow;
        const uint4* hb = (const uint4*)&whi[nrow * XPITCH];
        const uint4* lb = (const uint4*)&wlo[nrow * XPITCH];
        bh[ns].q[0] = hb[halfsel];
        bh[ns].q[1] = hb[halfsel + 2];
        bl[ns].q[0] = lb[halfsel];
        bl[ns].q[1] = lb[halfsel + 2];
    }
#pragma unroll
    for (int ms = 0; ms < 4; ++ms) {
        const int mrow = mOff + ms * 16 + lrow;
        const uint4* ha = (const uint4*)&xhi[mrow * XPITCH];
        const uint4* la = (const uint4*)&xlo[mrow * XPITCH];
        BFrag ah, al;
        ah.q[0] = ha[halfsel];
        ah.q[1] = ha[halfsel + 2];
        al.q[0] = la[halfsel];
        al.q[1] = la[halfsel + 2];
#pragma unroll
        for (int ns = 0; ns < 2; ++ns) {
            acc[ms][ns] = __builtin_amdgcn_wmma_f32_16x16x32_bf16(
                false, ah.v, false, bh[ns].v, (short)0, acc[ms][ns], false, false);
            acc[ms][ns] = __builtin_amdgcn_wmma_f32_16x16x32_bf16(
                false, ah.v, false, bl[ns].v, (short)0, acc[ms][ns], false, false);
            acc[ms][ns] = __builtin_amdgcn_wmma_f32_16x16x32_bf16(
                false, al.v, false, bh[ns].v, (short)0, acc[ms][ns], false, false);
        }
    }
}

// Epilogue: out = acc + bias + (t @ B) rank-16 exact fp32 update.
__device__ __forceinline__ void epilogue_store(const float* tEp, const float* bEp, const float* biasEp,
                                               v8f (&acc)[4][2], float* out,
                                               int blockM, int blockN, int mOff, int nOff,
                                               int lrow, int halfsel) {
    float bcol[2][16], bb[2];
#pragma unroll
    for (int ns = 0; ns < 2; ++ns) {
        const int ncol = nOff + ns * 16 + lrow;
        bb[ns] = biasEp[ncol];
#pragma unroll
        for (int r = 0; r < 16; ++r) bcol[ns][r] = bEp[r * BN + ncol];
    }
#pragma unroll
    for (int ms = 0; ms < 4; ++ms) {
#pragma unroll
        for (int i = 0; i < 8; ++i) {
            const int mrow = mOff + ms * 16 + i + 8 * halfsel;
            float tv[16];
            const float4* tr = (const float4*)&tEp[mrow * 16];
            *(float4*)&tv[0]  = tr[0];
            *(float4*)&tv[4]  = tr[1];
            *(float4*)&tv[8]  = tr[2];
            *(float4*)&tv[12] = tr[3];
#pragma unroll
            for (int ns = 0; ns < 2; ++ns) {
                float s = bb[ns];
#pragma unroll
                for (int r = 0; r < 16; ++r) s += tv[r] * bcol[ns][r];
                out[(size_t)(blockM + mrow) * FEAT_OUT + blockN + nOff + ns * 16 + lrow] =
                    acc[ms][ns][i] + s;
            }
        }
    }
}

// ---------------------------------------------------------------------------
// Kernel 1: t[m][r] = (ALPHA/R) * sum_k x[m][k] * A[k][r]
// ---------------------------------------------------------------------------
__global__ __launch_bounds__(256) void lora_t_kernel(const float* __restrict__ x,
                                                     const float* __restrict__ A,
                                                     float* __restrict__ t) {
    __shared__ float part[4][64][16];
    const int tid = threadIdx.x;
    const int ml  = tid >> 6;
    const int kl  = tid & 63;
    const int m   = blockIdx.x * 4 + ml;

    float acc[16];
#pragma unroll
    for (int r = 0; r < 16; ++r) acc[r] = 0.0f;

    const float* xr = x + (size_t)m * FEAT_IN;
    for (int k = kl; k < FEAT_IN; k += 64) {
        float xv = xr[k];
        float av[16];
        const float4* ar = (const float4*)(A + (size_t)k * 16);
        *(float4*)&av[0]  = ar[0];
        *(float4*)&av[4]  = ar[1];
        *(float4*)&av[8]  = ar[2];
        *(float4*)&av[12] = ar[3];
#pragma unroll
        for (int r = 0; r < 16; ++r) acc[r] += xv * av[r];
    }
#pragma unroll
    for (int r = 0; r < 16; ++r) part[ml][kl][r] = acc[r];
    __syncthreads();
    if (kl < 16) {
        float s = 0.0f;
        for (int j = 0; j < 64; ++j) s += part[ml][j][kl];
        t[(size_t)m * 16 + kl] = s * 2.0f;       // ALPHA / R
    }
}

// ---------------------------------------------------------------------------
// Kernel: streaming fp32 -> bf16 hi/lo split (one-time pass).
// ---------------------------------------------------------------------------
__global__ __launch_bounds__(256) void split_bf16_kernel(const float* __restrict__ src,
                                                         unsigned short* __restrict__ hi,
                                                         unsigned short* __restrict__ lo,
                                                         int n4) {
    const int idx = blockIdx.x * 256 + threadIdx.x;
    if (idx >= n4) return;
    float4 v = ((const float4*)src)[idx];
    float a[4] = {v.x, v.y, v.z, v.w};
    unsigned short h[4], l[4];
#pragma unroll
    for (int e = 0; e < 4; ++e) {
        h[e] = f2bf(a[e]);
        l[e] = f2bf(a[e] - bf2f(h[e]));
    }
    uint2 p;
    p.x = (unsigned)h[0] | ((unsigned)h[1] << 16);
    p.y = (unsigned)h[2] | ((unsigned)h[3] << 16);
    ((uint2*)hi)[idx] = p;
    p.x = (unsigned)l[0] | ((unsigned)l[1] << 16);
    p.y = (unsigned)l[2] | ((unsigned)l[3] << 16);
    ((uint2*)lo)[idx] = p;
}

// ---------------------------------------------------------------------------
// TDM descriptor issue: load one 128-row x 32-col bf16 tile into LDS with
// 4-DWORD padding every 16 DWORDs -> 80B LDS row pitch (== XPITCH ushorts).
// ---------------------------------------------------------------------------
__device__ __forceinline__ void tdm_issue(const unsigned short* gtile, unsigned ldsOff, unsigned rows) {
    const unsigned long long ga = (unsigned long long)(size_t)gtile;
    v4u g0 = { 1u,                                       // count=1
               ldsOff,                                   // lds_addr
               (unsigned)(ga & 0xFFFFFFFFu),             // global_addr lo
               (unsigned)((ga >> 32) & 0x01FFFFFFu) | (2u << 30) };  // addr hi | type=2
    v8i g1 = { (int)((1u << 16) | (1u << 20) | (3u << 22) | (3u << 25)), // 2B elems, pad en, 16DW interval, 4DW pad
               (int)(4096u << 16),                       // tensor_dim0 = 4096 (lo16)
               (int)(rows << 16),                        // dim0 hi=0 | tensor_dim1 lo16
               (int)((rows >> 16) | (32u << 16)),        // tensor_dim1 hi | tile_dim0 = 32
               128,                                      // tile_dim1 = 128, tile_dim2 = 0
               4096,                                     // tensor_dim0_stride lo32
               0, 0 };
    v4i g2 = { 1, 0, 0, 0 };                             // tensor_dim2 = 1 (unused)
    v4i g3 = { 0, 0, 0, 0 };
#if defined(__clang_major__) && (__clang_major__ >= 23)
    v8i g4 = { 0, 0, 0, 0, 0, 0, 0, 0 };
    __builtin_amdgcn_tensor_load_to_lds(g0, g1, g2, g3, g4, 0);
#else
    __builtin_amdgcn_tensor_load_to_lds(g0, g1, g2, g3, 0);
#endif
}

// ---------------------------------------------------------------------------
// Kernel 2a (TDM path): GEMM from pre-split bf16 hi/lo, TDM-staged, dbl-buffered.
// ---------------------------------------------------------------------------
__global__ __launch_bounds__(256) void lora_gemm_tdm_kernel(
    const unsigned short* __restrict__ xh, const unsigned short* __restrict__ xl,
    const unsigned short* __restrict__ wh, const unsigned short* __restrict__ wl,
    const float* __restrict__ bias, const float* __restrict__ loraB,
    const float* __restrict__ t, float* __restrict__ out) {

    __shared__ __align__(16) unsigned short xhT[2][BM * XPITCH];
    __shared__ __align__(16) unsigned short xlT[2][BM * XPITCH];
    __shared__ __align__(16) unsigned short whT[2][BN * XPITCH];
    __shared__ __align__(16) unsigned short wlT[2][BN * XPITCH];
    __shared__ __align__(16) float tEp[BM * 16];
    __shared__ __align__(16) float bEp[16 * BN];
    __shared__ float biasEp[BN];

    const int tid     = threadIdx.x;
    const int lane    = tid & 31;
    const int waveId  = tid >> 5;
    const int halfsel = lane >> 4;
    const int lrow    = lane & 15;
    const int blockN  = blockIdx.x * BN;
    const int blockM  = blockIdx.y * BM;
    const int mOff    = (waveId >> 2) * 64;
    const int nOff    = (waveId & 3) * 32;

    v8f acc[4][2];
#pragma unroll
    for (int ms = 0; ms < 4; ++ms)
#pragma unroll
        for (int ns = 0; ns < 2; ++ns)
#pragma unroll
            for (int e = 0; e < 8; ++e) acc[ms][ns][e] = 0.0f;

    // Stage epilogue operands up front (overlaps with first DMA).
    if (tid < BN) biasEp[tid] = bias[blockN + tid];
#pragma unroll
    for (int rep = 0; rep < 2; ++rep) {
        const int f4 = rep * 256 + tid;
        {
            const int row = f4 >> 2;
            const int col = (f4 & 3) << 2;
            *(float4*)&tEp[row * 16 + col] =
                *(const float4*)(t + (size_t)(blockM + row) * 16 + col);
        }
        {
            const int row = f4 >> 5;
            const int col = (f4 & 31) << 2;
            *(float4*)&bEp[row * BN + col] =
                *(const float4*)(loraB + (size_t)row * FEAT_OUT + blockN + col);
        }
    }

    const unsigned xhOff[2] = { (unsigned)(size_t)&xhT[0][0], (unsigned)(size_t)&xhT[1][0] };
    const unsigned xlOff[2] = { (unsigned)(size_t)&xlT[0][0], (unsigned)(size_t)&xlT[1][0] };
    const unsigned whOff[2] = { (unsigned)(size_t)&whT[0][0], (unsigned)(size_t)&whT[1][0] };
    const unsigned wlOff[2] = { (unsigned)(size_t)&wlT[0][0], (unsigned)(size_t)&wlT[1][0] };

    const unsigned short* xhBase = xh + (size_t)blockM * FEAT_IN;
    const unsigned short* xlBase = xl + (size_t)blockM * FEAT_IN;
    const unsigned short* whBase = wh + (size_t)blockN * FEAT_IN;
    const unsigned short* wlBase = wl + (size_t)blockN * FEAT_IN;

    if (waveId == 0) {   // prologue DMA: tile 0 into buffer 0
        tdm_issue(xhBase, xhOff[0], M_TOTAL);
        tdm_issue(xlBase, xlOff[0], M_TOTAL);
        tdm_issue(whBase, whOff[0], FEAT_OUT);
        tdm_issue(wlBase, wlOff[0], FEAT_OUT);
    }

    for (int ki = 0; ki < NKSTEPS; ++ki) {
        const int cur = ki & 1;
        if (waveId == 0) {
            if (ki + 1 < NKSTEPS) {      // kick off next tile, then wait for current
                const int kn = (ki + 1) * BK;
                tdm_issue(xhBase + kn, xhOff[cur ^ 1], M_TOTAL);
                tdm_issue(xlBase + kn, xlOff[cur ^ 1], M_TOTAL);
                tdm_issue(whBase + kn, whOff[cur ^ 1], FEAT_OUT);
                tdm_issue(wlBase + kn, wlOff[cur ^ 1], FEAT_OUT);
                __builtin_amdgcn_s_wait_tensorcnt((short)4);
            } else {
                __builtin_amdgcn_s_wait_tensorcnt((short)0);
            }
        }
        __syncthreads();   // tile `cur` visible to all waves
        wave_mma_step(&xhT[cur][0], &xlT[cur][0], &whT[cur][0], &wlT[cur][0],
                      mOff, nOff, lrow, halfsel, acc);
        __syncthreads();   // all reads done before buffer `cur` is overwritten
    }

    epilogue_store(tEp, bEp, biasEp, acc, out, blockM, blockN, mOff, nOff, lrow, halfsel);
}

// ---------------------------------------------------------------------------
// Kernel 2b (fallback): convert fp32 -> bf16 hi/lo in-kernel while staging.
// ---------------------------------------------------------------------------
__global__ __launch_bounds__(256) void lora_gemm_kernel(
    const float* __restrict__ x, const float* __restrict__ w,
    const float* __restrict__ bias, const float* __restrict__ loraB,
    const float* __restrict__ t, float* __restrict__ out) {

    __shared__ __align__(16) unsigned short xhi[BM * XPITCH];
    __shared__ __align__(16) unsigned short xlo[BM * XPITCH];
    __shared__ __align__(16) unsigned short whi[BN * XPITCH];
    __shared__ __align__(16) unsigned short wlo[BN * XPITCH];
    __shared__ __align__(16) float tEp[BM * 16];
    __shared__ __align__(16) float bEp[16 * BN];
    __shared__ float biasEp[BN];

    const int tid     = threadIdx.x;
    const int lane    = tid & 31;
    const int waveId  = tid >> 5;
    const int halfsel = lane >> 4;
    const int lrow    = lane & 15;
    const int blockN  = blockIdx.x * BN;
    const int blockM  = blockIdx.y * BM;
    const int mOff    = (waveId >> 2) * 64;
    const int nOff    = (waveId & 3) * 32;

    v8f acc[4][2];
#pragma unroll
    for (int ms = 0; ms < 4; ++ms)
#pragma unroll
        for (int ns = 0; ns < 2; ++ns)
#pragma unroll
            for (int e = 0; e < 8; ++e) acc[ms][ns][e] = 0.0f;

    if (tid < BN) biasEp[tid] = bias[blockN + tid];

    for (int k0 = 0; k0 < FEAT_IN; k0 += BK) {
        __syncthreads();
#pragma unroll
        for (int rep = 0; rep < 4; ++rep) {
            const int f4  = rep * 256 + tid;
            const int row = f4 >> 3;
            const int col = (f4 & 7) << 2;
            float4 xv = *(const float4*)(x + (size_t)(blockM + row) * FEAT_IN + k0 + col);
            float4 wv = *(const float4*)(w + (size_t)(blockN + row) * FEAT_IN + k0 + col);
            float vx[4] = {xv.x, xv.y, xv.z, xv.w};
            float vw[4] = {wv.x, wv.y, wv.z, wv.w};
            unsigned short hx[4], lx[4], hw_[4], lw_[4];
#pragma unroll
            for (int e = 0; e < 4; ++e) {
                hx[e]  = f2bf(vx[e]);
                lx[e]  = f2bf(vx[e] - bf2f(hx[e]));
                hw_[e] = f2bf(vw[e]);
                lw_[e] = f2bf(vw[e] - bf2f(hw_[e]));
            }
            uint2 p;
            p.x = (unsigned)hx[0] | ((unsigned)hx[1] << 16);
            p.y = (unsigned)hx[2] | ((unsigned)hx[3] << 16);
            *(uint2*)&xhi[row * XPITCH + col] = p;
            p.x = (unsigned)lx[0] | ((unsigned)lx[1] << 16);
            p.y = (unsigned)lx[2] | ((unsigned)lx[3] << 16);
            *(uint2*)&xlo[row * XPITCH + col] = p;
            p.x = (unsigned)hw_[0] | ((unsigned)hw_[1] << 16);
            p.y = (unsigned)hw_[2] | ((unsigned)hw_[3] << 16);
            *(uint2*)&whi[row * XPITCH + col] = p;
            p.x = (unsigned)lw_[0] | ((unsigned)lw_[1] << 16);
            p.y = (unsigned)lw_[2] | ((unsigned)lw_[3] << 16);
            *(uint2*)&wlo[row * XPITCH + col] = p;
        }
        if (k0 + BK < FEAT_IN) {
            __builtin_prefetch(x + (size_t)(blockM + (tid >> 1)) * FEAT_IN + k0 + BK, 0, 1);
            __builtin_prefetch(w + (size_t)(blockN + (tid >> 1)) * FEAT_IN + k0 + BK, 0, 1);
        }
        __syncthreads();
        wave_mma_step(xhi, xlo, whi, wlo, mOff, nOff, lrow, halfsel, acc);
    }

    __syncthreads();
#pragma unroll
    for (int rep = 0; rep < 2; ++rep) {
        const int f4 = rep * 256 + tid;
        {
            const int row = f4 >> 2;
            const int col = (f4 & 3) << 2;
            *(float4*)&tEp[row * 16 + col] =
                *(const float4*)(t + (size_t)(blockM + row) * 16 + col);
        }
        {
            const int row = f4 >> 5;
            const int col = (f4 & 31) << 2;
            *(float4*)&bEp[row * BN + col] =
                *(const float4*)(loraB + (size_t)row * FEAT_OUT + blockN + col);
        }
    }
    __syncthreads();
    epilogue_store(tEp, bEp, biasEp, acc, out, blockM, blockN, mOff, nOff, lrow, halfsel);
}

extern "C" void kernel_launch(void* const* d_in, const int* in_sizes, int n_in,
                              void* d_out, int out_size, void* d_ws, size_t ws_size,
                              hipStream_t stream) {
    (void)in_sizes; (void)n_in; (void)out_size;
    const float* x      = (const float*)d_in[0];
    const float* w      = (const float*)d_in[1];
    const float* b      = (const float*)d_in[2];
    const float* lora_a = (const float*)d_in[3];
    const float* lora_b = (const float*)d_in[4];
    float* out = (float*)d_out;

    const size_t T_BYTES  = (size_t)M_TOTAL * 16 * sizeof(float);       // 512 KB
    const size_t XS_BYTES = (size_t)M_TOTAL * FEAT_IN * 2;              // 64 MB per half
    const size_t WS_BYTES = (size_t)FEAT_OUT * FEAT_IN * 2;             // 32 MB per half
    const size_t NEED     = T_BYTES + 2 * XS_BYTES + 2 * WS_BYTES;      // ~192.5 MB

    float* t = (float*)d_ws;
    lora_t_kernel<<<dim3(M_TOTAL / 4), dim3(256), 0, stream>>>(x, lora_a, t);

    dim3 grid(FEAT_OUT / BN, M_TOTAL / BM);
    if (ws_size >= NEED) {
        char* base = (char*)d_ws + T_BYTES;
        unsigned short* xh = (unsigned short*)(base);
        unsigned short* xl = (unsigned short*)(base + XS_BYTES);
        unsigned short* wh = (unsigned short*)(base + 2 * XS_BYTES);
        unsigned short* wl = (unsigned short*)(base + 2 * XS_BYTES + WS_BYTES);
        const int n4x = M_TOTAL * FEAT_IN / 4;
        const int n4w = FEAT_OUT * FEAT_IN / 4;
        split_bf16_kernel<<<dim3(n4x / 256), dim3(256), 0, stream>>>(x, xh, xl, n4x);
        split_bf16_kernel<<<dim3(n4w / 256), dim3(256), 0, stream>>>(w, wh, wl, n4w);
        lora_gemm_tdm_kernel<<<grid, dim3(256), 0, stream>>>(xh, xl, wh, wl, b, lora_b, t, out);
    } else {
        lora_gemm_kernel<<<grid, dim3(256), 0, stream>>>(x, w, b, lora_b, t, out);
    }
}
